// DeformableCrossAttention_22445499089233
// MI455X (gfx1250) — matrix operand order repaired and checked by reading, not compiled, not run on hardware
//
#include <hip/hip_runtime.h>
#include <hip/hip_bf16.h>

typedef __attribute__((ext_vector_type(16))) _Float16 v16h;
typedef __attribute__((ext_vector_type(8)))  _Float16 h8;
typedef __attribute__((ext_vector_type(8)))  float    v8f;

#define HWSZ 3136
#define NCOLS 6272   // B*HW
#define DIMC 256

// ---------------- WMMA helpers (f16 in, f32 accum) ----------------
__device__ inline v8f wmma32(v16h a, v16h b, v8f c) {
  return __builtin_amdgcn_wmma_f32_16x16x32_f16(false, a, false, b, (short)0, c,
                                                false, false);
}
// A fragment: lane<16 -> row=lane, K 0..15 (contiguous);
//             lane>=16 -> row=lane-16, K 16..31.
// Caller passes p already offset by row*lda + (lane&16 ? 16 : 0).
__device__ inline v16h ld_a16(const _Float16* p) {
  union { v16h v; h8 h[2]; } u;
  u.h[0] = *(const h8*)(p);
  u.h[1] = *(const h8*)(p + 8);
  return u.v;
}
// B fragment: lane<16 -> col=lane, K rows kb..kb+15 (contiguous halves);
//             lane>=16 -> col=lane-16, K rows kb+16..kb+31.
__device__ inline v16h ld_b16(const _Float16* p) {
  union { v16h v; h8 h[2]; } u;
  u.h[0] = *(const h8*)(p);
  u.h[1] = *(const h8*)(p + 8);
  return u.v;
}
__device__ inline float gelu_tanh(float x) {
  return 0.5f * x * (1.0f + tanhf(0.7978845608028654f * (x + 0.044715f * x * x * x)));
}

// ---------------- conversion / prep kernels ----------------
__global__ __launch_bounds__(256) void f2h_kernel(const float* s, _Float16* d, int n) {
  int i = blockIdx.x * 256 + threadIdx.x;
  if (i < n) d[i] = (_Float16)s[i];
}
__global__ __launch_bounds__(256) void f2h_pad_kernel(const float* s, _Float16* d,
                                                      int n, int ntot) {
  int i = blockIdx.x * 256 + threadIdx.x;
  if (i < ntot) d[i] = (i < n) ? (_Float16)s[i] : (_Float16)0.0f;
}
// query_map (B,C,H,W) fp32 -> x_t[n][c] half (n = b*HW + hw)
__global__ __launch_bounds__(256) void xpose_cvt_kernel(const float* qm, _Float16* x_t) {
  int i = blockIdx.x * 256 + threadIdx.x;
  if (i < NCOLS * DIMC) {
    int ncol = i >> 8, c = i & 255;
    int b = ncol / HWSZ, hw = ncol % HWSZ;
    x_t[i] = (_Float16)qm[((size_t)b * DIMC + c) * HWSZ + hw];
  }
}
// continuous positional bias table: (2H-1)*(2W-1) x HEADS fp32
__global__ __launch_bounds__(256) void cpb_kernel(const float* w1, const float* b1,
                                                  const float* w2, const float* b2,
                                                  float* table) {
  int t = blockIdx.x * 256 + threadIdx.x;
  if (t < 111 * 111) {
    int iy = t / 111, ix = t % 111;
    float dy = (float)(iy - 55) / 55.0f;
    float dx = (float)(ix - 55) / 55.0f;
    float out[8];
#pragma unroll
    for (int h = 0; h < 8; ++h) out[h] = b2[h];
    for (int d = 0; d < 32; ++d) {
      float hd = dy * w1[d * 2] + dx * w1[d * 2 + 1] + b1[d];
      hd = fmaxf(hd, 0.0f);
#pragma unroll
      for (int h = 0; h < 8; ++h) out[h] += hd * w2[h * 32 + d];
    }
#pragma unroll
    for (int h = 0; h < 8; ++h) table[t * 8 + h] = out[h];
  }
}

// ---------------- generic wave GEMM: MT m-tiles per wave, B-frag reused MT x ----
// C[m,n] = sum_k A[m,k]*B[k,n];  A row-major (lda==K), B stored n-major: Bt[n][k].
// MODE 0: out = n-major half (ld LDOUT)            (q projection)
// MODE 1: +bias, GELU, n-major half (ld LDOUT)     (offset hidden)
// MODE 2: +bias, fp32 (B,18,HW), guard m<18        (offset out)
// MODE 3: +bias, fp32 (B,256,HW)                   (final out-proj)
template <int MODE, int K, int LDOUT, int MT>
__global__ __launch_bounds__(256) void gemm_kernel(const _Float16* __restrict__ A,
                                                   const _Float16* __restrict__ Bt,
                                                   const float* __restrict__ bias,
                                                   void* __restrict__ out, int mtiles) {
  int warp = threadIdx.x >> 5, lane = threadIdx.x & 31;
  int id = blockIdx.x * 8 + warp;
  int ngroups = mtiles / MT;
  int m0 = (id % ngroups) * MT * 16;
  int n0 = (id / ngroups) * 16;
  v8f acc[MT];
#pragma unroll
  for (int j = 0; j < MT; ++j)
    acc[j] = (v8f){0.f, 0.f, 0.f, 0.f, 0.f, 0.f, 0.f, 0.f};
  const _Float16* arow = A + (size_t)(m0 + (lane & 15)) * K + ((lane & 16) ? 16 : 0);
  const _Float16* brow = Bt + (size_t)(n0 + (lane & 15)) * K + ((lane & 16) ? 16 : 0);
  for (int kb = 0; kb < K; kb += 32) {
    v16h bf = ld_b16(brow + kb);
#pragma unroll
    for (int j = 0; j < MT; ++j)
      acc[j] = wmma32(ld_a16(arow + (size_t)j * 16 * K + kb), bf, acc[j]);
  }
  int nl = n0 + (lane & 15);
#pragma unroll
  for (int j = 0; j < MT; ++j) {
    int mb = m0 + j * 16 + ((lane & 16) ? 8 : 0);
    if (MODE == 0) {
      h8 hv;
#pragma unroll
      for (int r = 0; r < 8; ++r) hv[r] = (_Float16)acc[j][r];
      *(h8*)((_Float16*)out + (size_t)nl * LDOUT + mb) = hv;
    } else if (MODE == 1) {
      h8 hv;
#pragma unroll
      for (int r = 0; r < 8; ++r) hv[r] = (_Float16)gelu_tanh(acc[j][r] + bias[mb + r]);
      *(h8*)((_Float16*)out + (size_t)nl * LDOUT + mb) = hv;
    } else if (MODE == 2) {
      int b = nl / HWSZ, hw = nl % HWSZ;
#pragma unroll
      for (int r = 0; r < 8; ++r) {
        int m = mb + r;
        if (m < 18)
          ((float*)out)[((size_t)b * 18 + m) * HWSZ + hw] = acc[j][r] + bias[m];
      }
    } else {
      int b = nl / HWSZ, hw = nl % HWSZ;
#pragma unroll
      for (int r = 0; r < 8; ++r) {
        int m = mb + r;
        ((float*)out)[((size_t)b * DIMC + m) * HWSZ + hw] = acc[j][r] + bias[m];
      }
    }
  }
}

// ---------------- fused deformable sampling + K/V projection ----------------
// Each block owns a 16-column tile. Samples kv (2304 x 16) into LDS in 3 chunks
// of 768 K, and accumulates the Wk/Wv GEMMs (32 output m-tiles, 4 per wave).
// K written token-major: k_t[n][c].  V written channel-major: v_ch[b][c][hw].
__global__ __launch_bounds__(256) void sample_kv_kernel(
    const float* __restrict__ kv_map, const float* __restrict__ off_f,
    const _Float16* __restrict__ Wk_h, const _Float16* __restrict__ Wv_h,
    _Float16* __restrict__ k_t, _Float16* __restrict__ v_ch) {
  __shared__ __align__(16) _Float16 b_sh[16][768];
  __shared__ float s_w[16][9][4];
  __shared__ int s_ix[16][9][4];
  int t = threadIdx.x;
  int n0 = blockIdx.x * 16;

  if (t < 144) {  // bilinear weights for (n, o)
    int n = t / 9, o = t % 9;
    int ncol = n0 + n;
    int b = ncol / HWSZ, hw = ncol % HWSZ;
    int hh = hw / 56, ww = hw % 56;
    float ox = off_f[((size_t)b * 18 + o) * HWSZ + hw];
    float oy = off_f[((size_t)b * 18 + 9 + o) * HWSZ + hw];
    float gx = (-1.0f + 2.0f * ww / 55.0f) + ox * 0.1f;
    float gy = (-1.0f + 2.0f * hh / 55.0f) + oy * 0.1f;
    float px = fminf(fmaxf((gx + 1.0f) * 0.5f * 55.0f, 0.0f), 55.0f);
    float py = fminf(fmaxf((gy + 1.0f) * 0.5f * 55.0f, 0.0f), 55.0f);
    float fx = floorf(px), fy = floorf(py);
    int x0 = (int)fx, y0 = (int)fy;
    int x1 = x0 + 1 > 55 ? 55 : x0 + 1;
    int y1 = y0 + 1 > 55 ? 55 : y0 + 1;
    float wx = px - fx, wy = py - fy;
    s_w[n][o][0] = (1.f - wx) * (1.f - wy);
    s_w[n][o][1] = wx * (1.f - wy);
    s_w[n][o][2] = (1.f - wx) * wy;
    s_w[n][o][3] = wx * wy;
    s_ix[n][o][0] = y0 * 56 + x0;
    s_ix[n][o][1] = y0 * 56 + x1;
    s_ix[n][o][2] = y1 * 56 + x0;
    s_ix[n][o][3] = y1 * 56 + x1;
  }
  __syncthreads();

  int warp = t >> 5, lane = t & 31;
  v8f acc[4];
#pragma unroll
  for (int j = 0; j < 4; ++j) acc[j] = (v8f){0.f, 0.f, 0.f, 0.f, 0.f, 0.f, 0.f, 0.f};

  int nfill = t >> 4, c0 = t & 15;
  int ncolf = n0 + nfill;
  const float* kvb = kv_map + (size_t)(ncolf / HWSZ) * DIMC * HWSZ;

  for (int cc = 0; cc < 3; ++cc) {
    if (cc) __syncthreads();  // previous chunk fully consumed
    for (int k = c0; k < 768; k += 16) {
      int kg = cc * 768 + k;
      int o = kg >> 8, c = kg & 255;
      const float* pc = kvb + (size_t)c * HWSZ;
      float v = s_w[nfill][o][0] * pc[s_ix[nfill][o][0]] +
                s_w[nfill][o][1] * pc[s_ix[nfill][o][1]] +
                s_w[nfill][o][2] * pc[s_ix[nfill][o][2]] +
                s_w[nfill][o][3] * pc[s_ix[nfill][o][3]];
      b_sh[nfill][k] = (_Float16)v;
    }
    __syncthreads();
#pragma unroll
    for (int j = 0; j < 4; ++j) {
      int tt = warp + j * 8;
      const _Float16* W = (tt >= 16) ? Wv_h : Wk_h;
      int m0 = (tt & 15) * 16;
      const _Float16* arow =
          W + (size_t)(m0 + (lane & 15)) * 2304 + cc * 768 + ((lane & 16) ? 16 : 0);
      const _Float16* brow = &b_sh[lane & 15][(lane & 16) ? 16 : 0];
      for (int kb = 0; kb < 768; kb += 32)
        acc[j] = wmma32(ld_a16(arow + kb), ld_b16(brow + kb), acc[j]);
    }
  }

  int nl = n0 + (lane & 15);
  int bcol = nl / HWSZ, hwcol = nl % HWSZ;
#pragma unroll
  for (int j = 0; j < 4; ++j) {
    int tt = warp + j * 8;
    int mb = (tt & 15) * 16 + ((lane & 16) ? 8 : 0);
    if (tt < 16) {  // K: token-major, contiguous along channels
      h8 hv;
#pragma unroll
      for (int r = 0; r < 8; ++r) hv[r] = (_Float16)acc[j][r];
      *(h8*)&k_t[(size_t)nl * DIMC + mb] = hv;
    } else {  // V: channel-major
#pragma unroll
      for (int r = 0; r < 8; ++r)
        v_ch[((size_t)bcol * DIMC + mb + r) * HWSZ + hwcol] = (_Float16)acc[j][r];
    }
  }
}

// ---------------- flash attention with CPB bias ----------------
// One wave per (b, head, 16-row i-tile); 64 key columns per iteration to
// amortize the cross-lane softmax reductions over 2x more score elements.
// Q/K token-major (ld 256), V channel-major.
__global__ __launch_bounds__(256) void attn_kernel(const _Float16* __restrict__ q_t,
                                                   const _Float16* __restrict__ k_t,
                                                   const _Float16* __restrict__ v_ch,
                                                   const float* __restrict__ table,
                                                   _Float16* __restrict__ attn_t) {
  __shared__ __align__(16) _Float16 p_sh[8][16][64];
  int warp = threadIdx.x >> 5, lane = threadIdx.x & 31;
  int wid = blockIdx.x * 8 + warp;  // 0..3135
  int bh = wid / 196, it = wid % 196;
  int b = bh >> 3, h = bh & 7;
  int i0 = it * 16;
  int nbase = b * HWSZ;
  const float scale = 0.17677669529663687f;  // 32^-0.5
  int col = lane & 15;
  int rbase = (lane & 16) ? 8 : 0;
  int koff = (lane & 16) ? 16 : 0;   // B-frag K offset
  int aoff = (lane & 16) ? 16 : 0;   // A-frag K offset

  v16h qa = ld_a16(q_t + (size_t)(nbase + i0 + col) * DIMC + h * 32 + aoff);

  int yi[8], xi[8];
#pragma unroll
  for (int r = 0; r < 8; ++r) {
    int m = i0 + r + rbase;
    yi[r] = m / 56;
    xi[r] = m % 56;
  }
  float mrun[8], lrun[8];
#pragma unroll
  for (int r = 0; r < 8; ++r) { mrun[r] = -3.0e38f; lrun[r] = 0.0f; }
  v8f o0 = {0.f, 0.f, 0.f, 0.f, 0.f, 0.f, 0.f, 0.f};
  v8f o1 = o0;
  v8f zero = o0;

  for (int j0 = 0; j0 < HWSZ; j0 += 64) {
    v8f s[4];
#pragma unroll
    for (int u = 0; u < 4; ++u) {
      v16h bk = ld_b16(k_t + (size_t)(nbase + j0 + u * 16 + col) * DIMC + h * 32 + koff);
      s[u] = wmma32(qa, bk, zero);
    }
    int jy[4], jx[4];
#pragma unroll
    for (int u = 0; u < 4; ++u) {
      int j = j0 + u * 16 + col;
      jy[u] = j / 56;
      jx[u] = j % 56;
    }
#pragma unroll
    for (int r = 0; r < 8; ++r) {
      float sv[4];
#pragma unroll
      for (int u = 0; u < 4; ++u) {
        int idx = (yi[r] - jy[u] + 55) * 111 + (xi[r] - jx[u] + 55);
        sv[u] = s[u][r] * scale + table[idx * 8 + h];
      }
      float tmax = fmaxf(fmaxf(sv[0], sv[1]), fmaxf(sv[2], sv[3]));
#pragma unroll
      for (int off = 1; off < 16; off <<= 1)
        tmax = fmaxf(tmax, __shfl_xor(tmax, off, 32));
      float mn = fmaxf(mrun[r], tmax);
      float corr = __expf(mrun[r] - mn);
      float rs = 0.f;
#pragma unroll
      for (int u = 0; u < 4; ++u) {
        float pv = __expf(sv[u] - mn);
        rs += pv;
        p_sh[warp][r + rbase][u * 16 + col] = (_Float16)pv;
      }
#pragma unroll
      for (int off = 1; off < 16; off <<= 1) rs += __shfl_xor(rs, off, 32);
      lrun[r] = lrun[r] * corr + rs;
      mrun[r] = mn;
      o0[r] *= corr;
      o1[r] *= corr;
    }
    __builtin_amdgcn_wave_barrier();
    // P (C-layout) -> A fragments via per-wave LDS tile: K = 64 keys
    v16h pa0 = ld_a16(&p_sh[warp][col][aoff]);
    v16h pa1 = ld_a16(&p_sh[warp][col][32 + aoff]);
    const _Float16* vb0 = v_ch + ((size_t)b * DIMC + h * 32 + col) * HWSZ + j0 + koff;
    const _Float16* vb1 = vb0 + (size_t)16 * HWSZ;
    o0 = wmma32(pa0, ld_b16(vb0), o0);
    o0 = wmma32(pa1, ld_b16(vb0 + 32), o0);
    o1 = wmma32(pa0, ld_b16(vb1), o1);
    o1 = wmma32(pa1, ld_b16(vb1 + 32), o1);
    __builtin_amdgcn_wave_barrier();
  }
#pragma unroll
  for (int r = 0; r < 8; ++r) {
    float inv = 1.0f / lrun[r];
    size_t rowb = (size_t)(nbase + i0 + r + rbase) * DIMC + h * 32;
    attn_t[rowb + col] = (_Float16)(o0[r] * inv);
    attn_t[rowb + 16 + col] = (_Float16)(o1[r] * inv);
  }
}

// ---------------- host launch ----------------
extern "C" void kernel_launch(void* const* d_in, const int* in_sizes, int n_in,
                              void* d_out, int out_size, void* d_ws, size_t ws_size,
                              hipStream_t stream) {
  const float* query_map = (const float*)d_in[0];
  const float* kv_map = (const float*)d_in[1];
  const float* Wq = (const float*)d_in[2];
  const float* Wk = (const float*)d_in[3];
  const float* Wv = (const float*)d_in[4];
  const float* Woff1 = (const float*)d_in[5];
  const float* boff1 = (const float*)d_in[6];
  const float* Woff2 = (const float*)d_in[7];
  const float* boff2 = (const float*)d_in[8];
  const float* cw1 = (const float*)d_in[9];
  const float* cb1 = (const float*)d_in[10];
  const float* cw2 = (const float*)d_in[11];
  const float* cb2 = (const float*)d_in[12];
  const float* Wout = (const float*)d_in[13];
  const float* bout = (const float*)d_in[14];
  float* out = (float*)d_out;

  char* ws = (char*)d_ws;
  size_t o = 0;
  _Float16* Wq_h = (_Float16*)(ws + o); o += 131072;           // 256*256*2
  _Float16* Wk_h = (_Float16*)(ws + o); o += 1179648;          // 256*2304*2
  _Float16* Wv_h = (_Float16*)(ws + o); o += 1179648;
  _Float16* W1_h = (_Float16*)(ws + o); o += 32768;            // 64*256*2
  _Float16* W2_h = (_Float16*)(ws + o); o += 4096;             // 32*64*2 (padded)
  _Float16* Wo_h = (_Float16*)(ws + o); o += 131072;
  _Float16* x_t = (_Float16*)(ws + o); o += 3211264;           // 6272*256*2
  _Float16* q_t = (_Float16*)(ws + o); o += 3211264;
  _Float16* t1_t = (_Float16*)(ws + o); o += 802816;           // 6272*64*2
  float* off_f = (float*)(ws + o); o += 451584;                // 2*18*3136*4
  _Float16* k_t = (_Float16*)(ws + o); o += 3211264;
  _Float16* v_ch = (_Float16*)(ws + o); o += 3211264;
  float* table = (float*)(ws + o); o += 394496;                // 12321*8*4
  _Float16* attn_t = (_Float16*)(ws + o); o += 3211264;
  (void)ws_size; (void)n_in; (void)in_sizes; (void)out_size;

  // weight conversions + prep
  f2h_kernel<<<256, 256, 0, stream>>>(Wq, Wq_h, 65536);
  f2h_kernel<<<2304, 256, 0, stream>>>(Wk, Wk_h, 589824);
  f2h_kernel<<<2304, 256, 0, stream>>>(Wv, Wv_h, 589824);
  f2h_kernel<<<64, 256, 0, stream>>>(Woff1, W1_h, 16384);
  f2h_pad_kernel<<<8, 256, 0, stream>>>(Woff2, W2_h, 1152, 2048);
  f2h_kernel<<<256, 256, 0, stream>>>(Wout, Wo_h, 65536);
  xpose_cvt_kernel<<<6272, 256, 0, stream>>>(query_map, x_t);
  cpb_kernel<<<49, 256, 0, stream>>>(cw1, cb1, cw2, cb2, table);

  // q = Wq * x               (M=256 -> 4 groups of 4 m-tiles; 392 n-tiles)
  gemm_kernel<0, 256, 256, 4><<<196, 256, 0, stream>>>(Wq_h, x_t, nullptr, q_t, 16);
  // t1 = gelu(Woff1*q + b1)  (M=64 -> 1 group of 4)
  gemm_kernel<1, 256, 64, 4><<<49, 256, 0, stream>>>(W1_h, q_t, boff1, t1_t, 4);
  // off = Woff2*t1 + b2      (M=18 padded to 32 -> 1 group of 2)
  gemm_kernel<2, 64, 0, 2><<<49, 256, 0, stream>>>(W2_h, t1_t, boff2, off_f, 2);
  // fused deformable sampling + K/V projections
  sample_kv_kernel<<<392, 256, 0, stream>>>(kv_map, off_f, Wk_h, Wv_h, k_t, v_ch);
  // flash attention with CPB bias
  attn_kernel<<<392, 256, 0, stream>>>(q_t, k_t, v_ch, table, attn_t);
  // final projection
  gemm_kernel<3, 256, 0, 4><<<196, 256, 0, stream>>>(Wo_h, attn_t, bout, out, 16);
}
